// NTM_S4D_31164282700621
// MI455X (gfx1250) — matrix operand order repaired and checked by reading, not compiled, not run on hardware
//
#include <hip/hip_runtime.h>
#include <math.h>

// ---------------- problem constants ----------------
#define BB    32      // batch
#define LL    256     // seq len
#define EXT   64
#define HH    256     // hidden
#define MM    64      // memory word
#define NMEM  4096    // memory rows
#define NST   32      // S4D states (== wave32 lanes!)
#define OUTD  64
#define WOUT  198     // 3*M + 6
#define ROUT  70      // M + 6

#define WLDS_STRIDE 65   // 64 dwords + 1 pad dword (TDM pad_interval=5, pad_amount=0)

typedef __attribute__((ext_vector_type(16))) _Float16 v16h;
typedef __attribute__((ext_vector_type(8)))  float    v8f;
typedef __attribute__((ext_vector_type(4)))  unsigned int u32x4;
typedef __attribute__((ext_vector_type(8)))  unsigned int u32x8;

// ---------------- helpers ----------------
__device__ __forceinline__ float wave_sum32(float v) {
    #pragma unroll
    for (int off = 16; off > 0; off >>= 1) v += __shfl_xor(v, off, 32);
    return v;
}
__device__ __forceinline__ float sigm(float x)  { return 1.0f / (1.0f + expf(-x)); }
__device__ __forceinline__ float softp(float x) { return (x > 20.0f) ? x : log1pf(expf(x)); }

// =====================================================================
// K1: encoder GEMM  u[(b*L+l)*H + h] = xin @ enc_W + enc_b
//     xin[row][k] = (k<64) ? x[b][l][k] : r0[b][k-64],  row = b*L + l
//     Block (256 thr, 8 waves) computes a 128-row x 64-col slab.
//     B panel (enc_W slab, 128x64 f32) staged into LDS by the Tensor Data
//     Mover with +1-DWORD row padding (conflict-free ds reads), then each
//     wave runs 4 K-chunks x 4 N-tiles = 16 v_wmma_f32_16x16x32_f16.
// =====================================================================
__global__ __launch_bounds__(256)
void enc_gemm_kernel(const float* __restrict__ x, const float* __restrict__ r0,
                     const float* __restrict__ enc_W, const float* __restrict__ enc_b,
                     float* __restrict__ u) {
    __shared__ float Wl[128 * WLDS_STRIDE];   // 33.3 KB of the 320KB WGP pool

    const int lane = threadIdx.x & 31;
    const int wv   = threadIdx.x >> 5;        // wave 0..7 -> row tile
    const int g    = lane >> 4;               // half-wave
    const int mr   = lane & 15;
    const int hbase = blockIdx.y * 64;        // column slab base

    // ---- TDM: stage enc_W[0:128, hbase:hbase+64] (f32) into LDS ----
    // D# per CDNA5 ISA. 2D tile 64x128, data_size=4B,
    // pad_enable: +1 DWORD every 64 DWORDs -> LDS row stride 65 DWORDs.
    if (threadIdx.x < 32) {
        const unsigned long long ga = (unsigned long long)(const void*)(enc_W + hbase);
        u32x4 g0;
        g0[0] = 1u;                                        // count=1 (valid D#)
        g0[1] = (unsigned)(unsigned long long)(const void*)&Wl[0]; // lds_addr (low 32b of flat shared = LDS offset)
        g0[2] = (unsigned)(ga & 0xffffffffu);              // global_addr[31:0] (tile start)
        g0[3] = (unsigned)((ga >> 32) & 0x1ffffffu) | (2u << 30); // addr[56:32] | type=2
        u32x8 g1;
        g1[0] = (2u << 16) | (1u << 20) | (5u << 22);      // data_size=4B, pad_enable, pad_interval=64dw
        g1[1] = (256u << 16);                              // tensor_dim0 = 256 (row length)
        g1[2] = (128u << 16);                              // tensor_dim1 = 128
        g1[3] = (64u  << 16);                              // tile_dim0   = 64
        g1[4] = 128u;                                      // tile_dim1   = 128
        g1[5] = 256u;                                      // tensor_dim0_stride = 256
        g1[6] = 0u;
        g1[7] = 0u;
        asm volatile("tensor_load_to_lds %0, %1" :: "s"(g0), "s"(g1) : "memory");
        __builtin_amdgcn_s_wait_tensorcnt(0);              // s_wait_tensorcnt 0
    }
    __syncthreads();

    // accumulators: 4 column tiles; C/D layout: lane -> col n=mr, VGPR j -> row m=j+8g
    v8f acc[4];
    #pragma unroll
    for (int nt = 0; nt < 4; ++nt) {
        const float bias = enc_b[hbase + nt * 16 + mr];
        #pragma unroll
        for (int j = 0; j < 8; ++j) acc[nt][j] = bias;
    }

    // A source row for this lane: m = mr of row tile wv
    const int gm = blockIdx.x * 128 + wv * 16 + mr;   // global row = b*L + l
    const int bb = gm >> 8;                           // L = 256
    const int ll = gm & 255;
    const float* xrow = x  + ((size_t)bb * LL + ll) * EXT;
    const float* rrow = r0 + (size_t)bb * MM;

    #pragma unroll
    for (int kc = 0; kc < 4; ++kc) {
        const int kk = kc * 32;
        // ---- A fragment (16x32 f16): halves 0..7 -> K=kk+8g+e, 8..15 -> K=kk+16+8g+e
        const float* src = (kk < EXT) ? (xrow + kk) : (rrow + (kk - EXT));
        v16h a;
        #pragma unroll
        for (int e = 0; e < 8; ++e) a[e]     = (_Float16)src[g * 8 + e];
        #pragma unroll
        for (int e = 0; e < 8; ++e) a[8 + e] = (_Float16)src[16 + g * 8 + e];

        // ---- 4 column tiles share this A fragment
        #pragma unroll
        for (int nt = 0; nt < 4; ++nt) {
            // B fragment (32x16 f16): lane holds column N=mr, halves 0..7 -> K=kk+8g+e, 8..15 -> K=kk+16+8g+e
            const float* wrow = &Wl[(kk + g * 8) * WLDS_STRIDE + nt * 16 + mr];
            v16h bf;
            #pragma unroll
            for (int e = 0; e < 16; ++e) bf[e] = (_Float16)wrow[((e & 7) + (e >> 3) * 16) * WLDS_STRIDE];
            acc[nt] = __builtin_amdgcn_wmma_f32_16x16x32_f16(false, a, false, bf,
                                                             (short)0, acc[nt], false, false);
        }
    }

    // store D: row m = j + 8g of tile, col = mr of column tile
    #pragma unroll
    for (int nt = 0; nt < 4; ++nt) {
        #pragma unroll
        for (int j = 0; j < 8; ++j) {
            const int row = blockIdx.x * 128 + wv * 16 + j + 8 * g;
            u[(size_t)row * HH + hbase + nt * 16 + mr] = acc[nt][j];
        }
    }
}

// =====================================================================
// K2: S4D scan. One wave per (b,h); lane = state index n (NST==32).
//     Fuses y into running mean -> xm[b][h] directly.
// =====================================================================
__global__ __launch_bounds__(256)
void scan_kernel(const float* __restrict__ u,
                 const float* __restrict__ s_real0, const float* __restrict__ s_imag0,
                 const float* __restrict__ log_dt, const float* __restrict__ log_A_real,
                 const float* __restrict__ A_imag, const float* __restrict__ C_real,
                 const float* __restrict__ C_imag, const float* __restrict__ Dp,
                 float* __restrict__ xm) {
    const int wid  = (blockIdx.x * blockDim.x + threadIdx.x) >> 5; // 0..B*H-1
    const int lane = threadIdx.x & 31;                             // state n
    const int b = wid / HH;
    const int h = wid % HH;
    const int hn = h * NST + lane;

    const float dt = expf(log_dt[h]);
    const float Ar = -expf(log_A_real[hn]);
    const float Ai = A_imag[hn];
    const float er  = expf(dt * Ar);
    const float dAr = er * cosf(dt * Ai);
    const float dAi = er * sinf(dt * Ai);
    const float den  = Ar * Ar + Ai * Ai;
    const float nr   = dAr - 1.0f;
    const float dBr  = (nr * Ar + dAi * Ai) / den;
    const float dBi  = (dAi * Ar - nr * Ai) / den;
    const float Cr = C_real[hn], Ci = C_imag[hn];
    const float Dh = Dp[h];

    float sr = s_real0[((size_t)b * HH + h) * NST + lane];
    float si = s_imag0[((size_t)b * HH + h) * NST + lane];
    float acc = 0.0f;

    const float* ub = u + (size_t)b * LL * HH + h;
    for (int l = 0; l < LL; ++l) {
        const float uk = ub[(size_t)l * HH];
        const float nsr = dAr * sr - dAi * si + dBr * uk;
        const float nsi = dAr * si + dAi * sr + dBi * uk;
        sr = nsr; si = nsi;
        float t = Cr * sr - Ci * si;          // Re(C * s)
        t = wave_sum32(t);                    // sum over n
        acc += 2.0f * t + Dh * uk + uk;       // y_k + u_k
    }
    if (lane == 0) xm[b * HH + h] = acc * (1.0f / (float)LL);
}

// =====================================================================
// K3: xm projections (kept fp32: feeds softplus/sigmoid/softmax/pow gates)
// =====================================================================
__global__ __launch_bounds__(256)
void proj_kernel(const float* __restrict__ xm,
                 const float* __restrict__ wW, const float* __restrict__ wb,
                 const float* __restrict__ rW, const float* __restrict__ rb,
                 float* __restrict__ ow, float* __restrict__ orh) {
    const int b = blockIdx.x;
    const float* xv = xm + b * HH;
    for (int j = threadIdx.x; j < WOUT + ROUT; j += blockDim.x) {
        if (j < WOUT) {
            float acc = wb[j];
            for (int h = 0; h < HH; ++h) acc += xv[h] * wW[(size_t)h * WOUT + j];
            ow[b * WOUT + j] = acc;
        } else {
            const int jj = j - WOUT;
            float acc = rb[jj];
            for (int h = 0; h < HH; ++h) acc += xv[h] * rW[(size_t)h * ROUT + jj];
            orh[b * ROUT + jj] = acc;
        }
    }
}

// =====================================================================
// K4/K8: per-batch head params: {beta, g, s0..s2, gamma, ||k+eps||}
// =====================================================================
__global__ __launch_bounds__(64)
void split_kernel(const float* __restrict__ obuf, int ostride, float* __restrict__ params) {
    const int b = blockIdx.x, t = threadIdx.x;
    const float* o = obuf + (size_t)b * ostride;
    __shared__ float red[64];
    const float kk = o[t] + 1e-16f;
    red[t] = kk * kk;
    __syncthreads();
    for (int s2 = 32; s2 > 0; s2 >>= 1) {
        if (t < s2) red[t] += red[t + s2];
        __syncthreads();
    }
    if (t == 0) {
        const float normk = sqrtf(red[0]);
        const float beta  = softp(o[MM]);
        const float gg    = sigm(o[MM + 1]);
        const float a0 = o[MM + 2], a1 = o[MM + 3], a2 = o[MM + 4];
        const float mx = fmaxf(a0, fmaxf(a1, a2));
        const float e0 = expf(a0 - mx), e1 = expf(a1 - mx), e2 = expf(a2 - mx);
        const float inv = 1.0f / (e0 + e1 + e2);
        const float gam = 1.0f + softp(o[MM + 5]);
        float* p = params + b * 8;
        p[0] = beta; p[1] = gg; p[2] = e0 * inv; p[3] = e1 * inv; p[4] = e2 * inv;
        p[5] = gam;  p[6] = normk;
    }
}

// =====================================================================
// K5/K9: cosine-similarity logits. One wave per (b,n): coalesced row dot.
// =====================================================================
__global__ __launch_bounds__(256)
void addr_kernel(const float* __restrict__ memp, const float* __restrict__ kbuf, int kstride,
                 const float* __restrict__ params, float* __restrict__ logits) {
    const int wid  = (blockIdx.x * blockDim.x + threadIdx.x) >> 5;
    const int lane = threadIdx.x & 31;
    const int b = wid / NMEM;
    const int n = wid % NMEM;
    const float* row = memp + ((size_t)b * NMEM + n) * MM;
    const float* kv  = kbuf + (size_t)b * kstride;
    float dot = 0.0f, nrm = 0.0f;
    #pragma unroll
    for (int j = 0; j < 2; ++j) {
        const int m = lane + j * 32;
        const float mv = row[m] + 1e-16f;
        const float kk = kv[m]  + 1e-16f;
        dot += mv * kk;
        nrm += mv * mv;
    }
    dot = wave_sum32(dot);
    nrm = wave_sum32(nrm);
    if (lane == 0) {
        const float beta  = params[b * 8 + 0];
        const float normk = params[b * 8 + 6];
        logits[(size_t)b * NMEM + n] = beta * dot / (sqrtf(nrm) * normk + 1e-16f);
    }
}

// =====================================================================
// K6/K10: per-batch softmax + interpolation + circular shift + sharpen.
// =====================================================================
__global__ __launch_bounds__(256)
void combine_kernel(const float* __restrict__ logits, const float* __restrict__ w_prev,
                    const float* __restrict__ params, float* __restrict__ w_out) {
    const int b = blockIdx.x, t = threadIdx.x;
    __shared__ float sh[NMEM];
    __shared__ float red[256];
    const float gg  = params[b * 8 + 1];
    const float s0  = params[b * 8 + 2], s1 = params[b * 8 + 3], s2 = params[b * 8 + 4];
    const float gam = params[b * 8 + 5];
    const float* lg = logits + (size_t)b * NMEM;
    const float* wp0 = w_prev + (size_t)b * NMEM;

    float lmax = -3.4e38f;
    #pragma unroll
    for (int i = 0; i < NMEM / 256; ++i) lmax = fmaxf(lmax, lg[t + i * 256]);
    red[t] = lmax; __syncthreads();
    for (int s = 128; s > 0; s >>= 1) { if (t < s) red[t] = fmaxf(red[t], red[t + s]); __syncthreads(); }
    const float gmax = red[0]; __syncthreads();

    float lsum = 0.0f;
    #pragma unroll
    for (int i = 0; i < NMEM / 256; ++i) {
        const int n = t + i * 256;
        const float v = expf(lg[n] - gmax);
        sh[n] = v; lsum += v;
    }
    red[t] = lsum; __syncthreads();
    for (int s = 128; s > 0; s >>= 1) { if (t < s) red[t] += red[t + s]; __syncthreads(); }
    const float inv = 1.0f / red[0]; __syncthreads();

    #pragma unroll
    for (int i = 0; i < NMEM / 256; ++i) {
        const int n = t + i * 256;
        sh[n] = gg * (sh[n] * inv) + (1.0f - gg) * wp0[n];
    }
    __syncthreads();

    float wpv[NMEM / 256];
    float psum = 0.0f;
    #pragma unroll
    for (int i = 0; i < NMEM / 256; ++i) {
        const int n = t + i * 256;
        const float wt = sh[(n + NMEM - 1) & (NMEM - 1)] * s0 + sh[n] * s1 +
                         sh[(n + 1) & (NMEM - 1)] * s2;
        const float p = powf(wt, gam);
        wpv[i] = p; psum += p;
    }
    red[t] = psum; __syncthreads();
    for (int s = 128; s > 0; s >>= 1) { if (t < s) red[t] += red[t + s]; __syncthreads(); }
    const float invp = 1.0f / (red[0] + 1e-16f);
    #pragma unroll
    for (int i = 0; i < NMEM / 256; ++i) w_out[(size_t)b * NMEM + t + i * 256] = wpv[i] * invp;
}

// =====================================================================
// K7: mem2 = mem*(1 - w_w*e) + w_w*a   (float4 streaming, 33MB r + 33MB w)
// =====================================================================
__global__ __launch_bounds__(256)
void memupd_kernel(const float* __restrict__ memp, const float* __restrict__ w_w,
                   const float* __restrict__ ow, float* __restrict__ mem2) {
    const size_t i = (size_t)blockIdx.x * blockDim.x + threadIdx.x;
    const size_t total4 = (size_t)BB * NMEM * MM / 4;
    if (i >= total4) return;
    const size_t e = i * 4;
    const int b   = (int)(e / ((size_t)NMEM * MM));
    const int rem = (int)(e % ((size_t)NMEM * MM));
    const int n = rem / MM;
    const int m = rem % MM;
    const float w = w_w[(size_t)b * NMEM + n];
    const float* o = ow + (size_t)b * WOUT;
    const float4 mv = *(const float4*)(memp + e);
    float4 r;
    r.x = mv.x * (1.0f - w * sigm(o[MM + 6 + m + 0])) + w * o[2 * MM + 6 + m + 0];
    r.y = mv.y * (1.0f - w * sigm(o[MM + 6 + m + 1])) + w * o[2 * MM + 6 + m + 1];
    r.z = mv.z * (1.0f - w * sigm(o[MM + 6 + m + 2])) + w * o[2 * MM + 6 + m + 2];
    r.w = mv.w * (1.0f - w * sigm(o[MM + 6 + m + 3])) + w * o[2 * MM + 6 + m + 3];
    *(float4*)(mem2 + e) = r;
}

// =====================================================================
// K11: r[b][m] = sum_n w_r[b][n] * mem2[b][n][m]
// =====================================================================
__global__ __launch_bounds__(256)
void readred_kernel(const float* __restrict__ w_r, const float* __restrict__ mem2,
                    float* __restrict__ rvec) {
    const int b = blockIdx.x, t = threadIdx.x;
    const int m = t & (MM - 1);
    const int grp = t >> 6;
    const float* wb = w_r + (size_t)b * NMEM;
    const float* mb = mem2 + (size_t)b * NMEM * MM;
    float acc = 0.0f;
    for (int n = grp; n < NMEM; n += 4) {
        __builtin_prefetch(mb + (size_t)(n + 16) * MM + m, 0, 0); // global_prefetch_b8
        acc += wb[n] * mb[(size_t)n * MM + m];
    }
    __shared__ float sh[256];
    sh[t] = acc; __syncthreads();
    if (grp == 0) rvec[b * MM + m] = sh[m] + sh[64 + m] + sh[128 + m] + sh[192 + m];
}

// =====================================================================
// K12: out = concat(xm, r) @ dec_W + dec_b  via WMMA
//     rows = B = 32 (2 tiles), cols = 64 (4 tiles), K = 320 (10 chunks).
//     Single block, 8 waves: wave -> (rowtile, coltile), 10 WMMAs each.
// =====================================================================
__global__ __launch_bounds__(256)
void decode_kernel(const float* __restrict__ xm, const float* __restrict__ rvec,
                   const float* __restrict__ dec_W, const float* __restrict__ dec_b,
                   float* __restrict__ out) {
    const int lane = threadIdx.x & 31;
    const int wv   = threadIdx.x >> 5;
    const int rt   = wv >> 2;        // 0..1 row tile (batch)
    const int ct   = wv & 3;         // 0..3 col tile
    const int g    = lane >> 4;
    const int mr   = lane & 15;

    const float bias = dec_b[ct * 16 + mr];
    v8f acc;
    #pragma unroll
    for (int j = 0; j < 8; ++j) acc[j] = bias;

    const int brow = rt * 16 + mr;   // batch index for this lane's A row
    const float* xrow = xm   + (size_t)brow * HH;
    const float* rrow = rvec + (size_t)brow * MM;

    #pragma unroll
    for (int kc = 0; kc < 10; ++kc) {
        const int kk = kc * 32;
        const float* src = (kk < HH) ? (xrow + kk) : (rrow + (kk - HH));
        v16h a;
        #pragma unroll
        for (int e = 0; e < 8; ++e) a[e]     = (_Float16)src[g * 8 + e];
        #pragma unroll
        for (int e = 0; e < 8; ++e) a[8 + e] = (_Float16)src[16 + g * 8 + e];
        const float* wrow = dec_W + (size_t)(kk + g * 8) * OUTD + ct * 16 + mr;
        v16h bf;
        #pragma unroll
        for (int e = 0; e < 16; ++e) bf[e] = (_Float16)wrow[((e & 7) + (e >> 3) * 16) * OUTD];
        acc = __builtin_amdgcn_wmma_f32_16x16x32_f16(false, a, false, bf,
                                                     (short)0, acc, false, false);
    }

    #pragma unroll
    for (int j = 0; j < 8; ++j) {
        const int row = rt * 16 + j + 8 * g;
        out[(size_t)row * OUTD + ct * 16 + mr] = acc[j];
    }
}

// ---------------- launch ----------------
extern "C" void kernel_launch(void* const* d_in, const int* in_sizes, int n_in,
                              void* d_out, int out_size, void* d_ws, size_t ws_size,
                              hipStream_t stream) {
    const float* x       = (const float*)d_in[0];
    const float* r0      = (const float*)d_in[1];
    const float* s_real0 = (const float*)d_in[2];
    const float* s_imag0 = (const float*)d_in[3];
    const float* w_r0    = (const float*)d_in[4];
    const float* w_w0    = (const float*)d_in[5];
    const float* memp    = (const float*)d_in[6];
    const float* enc_W   = (const float*)d_in[7];
    const float* enc_b   = (const float*)d_in[8];
    const float* log_dt  = (const float*)d_in[9];
    const float* log_A   = (const float*)d_in[10];
    const float* A_imag  = (const float*)d_in[11];
    const float* C_real  = (const float*)d_in[12];
    const float* C_imag  = (const float*)d_in[13];
    const float* Dp      = (const float*)d_in[14];
    const float* dec_W   = (const float*)d_in[15];
    const float* dec_b   = (const float*)d_in[16];
    const float* read_W  = (const float*)d_in[17];
    const float* read_b  = (const float*)d_in[18];
    const float* write_W = (const float*)d_in[19];
    const float* write_b = (const float*)d_in[20];

    float* ws = (float*)d_ws;
    float* U    = ws;                          // B*L*H
    float* XM   = U    + (size_t)BB * LL * HH; // B*H
    float* OW   = XM   + (size_t)BB * HH;      // B*198
    float* ORH  = OW   + (size_t)BB * WOUT;    // B*70
    float* PW   = ORH  + (size_t)BB * ROUT;    // B*8
    float* PR   = PW   + (size_t)BB * 8;       // B*8
    float* LOG  = PR   + (size_t)BB * 8;       // B*NMEM
    float* WW   = LOG  + (size_t)BB * NMEM;    // B*NMEM
    float* WR   = WW   + (size_t)BB * NMEM;    // B*NMEM
    float* RV   = WR   + (size_t)BB * NMEM;    // B*M
    float* MEM2 = RV   + (size_t)BB * MM;      // B*NMEM*M

    // K1: encoder GEMM (TDM-staged B panel + WMMA)
    enc_gemm_kernel<<<dim3((BB * LL) / 128, HH / 64), 256, 0, stream>>>(x, r0, enc_W, enc_b, U);
    // K2: S4D scan -> xm
    scan_kernel<<<(BB * HH) / 8, 256, 0, stream>>>(U, s_real0, s_imag0, log_dt, log_A,
                                                   A_imag, C_real, C_imag, Dp, XM);
    // K3: projections
    proj_kernel<<<BB, 256, 0, stream>>>(XM, write_W, write_b, read_W, read_b, OW, ORH);
    // K4-K6: write addressing
    split_kernel<<<BB, 64, 0, stream>>>(OW, WOUT, PW);
    addr_kernel<<<(BB * NMEM) / 8, 256, 0, stream>>>(memp, OW, WOUT, PW, LOG);
    combine_kernel<<<BB, 256, 0, stream>>>(LOG, w_w0, PW, WW);
    // K7: memory update
    memupd_kernel<<<((size_t)BB * NMEM * MM / 4 + 255) / 256, 256, 0, stream>>>(memp, WW, OW, MEM2);
    // K8-K10: read addressing (on mem2)
    split_kernel<<<BB, 64, 0, stream>>>(ORH, ROUT, PR);
    addr_kernel<<<(BB * NMEM) / 8, 256, 0, stream>>>(MEM2, ORH, ROUT, PR, LOG);
    combine_kernel<<<BB, 256, 0, stream>>>(LOG, w_r0, PR, WR);
    // K11: read vector
    readred_kernel<<<BB, 256, 0, stream>>>(WR, MEM2, RV);
    // K12: decode (WMMA) -> d_out
    decode_kernel<<<1, 256, 0, stream>>>(XM, RV, dec_W, dec_b, (float*)d_out);
}
